// DiffPool_48696339202114
// MI455X (gfx1250) — compile-verified, matrix operand order checked by reading, not run on hardware
//
#include <hip/hip_runtime.h>
#include <hip/hip_bf16.h>

// ---------------------------------------------------------------------------
// DiffPool forward for MI455X (gfx1250, wave32, WMMA).
//
// Exactness note: softmax over k IDENTICAL logits (proto broadcast to k rows)
// is exactly uniform 1/k; argmax of identical values is index 0. Hence
//   A            == 1/k everywhere
//   new_x[j,:]   == colsum(h)/k  (same row, k times)
//   edge_mask    == all false,  new_edge_index == 0,  new_edge_attr == 0
//   new_node_types[j] == colsum(node_types)/k
// The assignment MLP / degree centrality / proto are mathematically dead.
// Live compute = EfficientGraphConv (node MLP, edge MLP, gather*mul,
// scatter-add, LayerNorm) feeding colsum(h).
// ---------------------------------------------------------------------------

typedef __attribute__((ext_vector_type(16))) _Float16 v16h;
typedef __attribute__((ext_vector_type(8)))  float    v8f;
typedef __attribute__((ext_vector_type(4)))  float    v4f;

#define N_NODES 8192
#define N_EDGES 524288
#define NDIM    32
#define EDIM    16
#define KCL     4096

// Output layout (floats, concatenated in reference return order)
#define OFF_NEWX   0            // 4096*32      = 131072
#define OFF_EIDX   131072       // 2*E          = 1048576
#define OFF_EATTR  1179648      // E*16         = 8388608
#define OFF_A      9568256      // N*K          = 33554432
#define OFF_NNT    43122688     // K            = 4096
#define OFF_MASK   43126784     // E            = 524288

#define WMMA16(A,B,C) \
  __builtin_amdgcn_wmma_f32_16x16x32_f16(false,(A),false,(B),(short)0,(C),false,false)

__device__ __forceinline__ float lrelu(float v){ return v > 0.f ? v : 0.1f * v; }

// ISA 7.12.2: 16-bit A 16x32 layout. element e of lane L maps to
// K = e+8*hi (e<8) or 16+(e-8)+8*hi (e>=8), hi = L>=16.
__device__ __forceinline__ int kmap(int e, int hi){
  return (e < 8) ? (e + 8*hi) : (16 + (e - 8) + 8*hi);
}

// A fragment (M=16 rows x K=32) from an f32 LDS tile, zero-padded past kvalid.
__device__ __forceinline__ v16h frag_a_f32(const float* __restrict__ T, int stride,
                                           int kbase, int kvalid, int lane){
  int r = lane & 15, hi = lane >> 4;
  v16h a;
#pragma unroll
  for (int e = 0; e < 16; ++e){
    int K = kbase + kmap(e, hi);
    a[e] = (K < kvalid) ? (_Float16)T[r*stride + K] : (_Float16)0.f;
  }
  return a;
}

// A fragment from an f16 LDS tile (hidden activations).
__device__ __forceinline__ v16h frag_a_f16(const _Float16* __restrict__ T, int stride,
                                           int kbase, int lane){
  int r = lane & 15, hi = lane >> 4;
  v16h a;
#pragma unroll
  for (int e = 0; e < 16; ++e)
    a[e] = T[r*stride + kbase + kmap(e, hi)];
  return a;
}

// B fragment (K=32 x N=16) for B = W^T where W is [dout, din] row-major in LDS.
__device__ __forceinline__ v16h frag_b_w(const float* __restrict__ W, int din,
                                         int d0, int kbase, int kvalid, int lane){
  int n = lane & 15, hi = lane >> 4;
  v16h b;
#pragma unroll
  for (int e = 0; e < 16; ++e){
    int K = kbase + kmap(e, hi);
    b[e] = (K < kvalid) ? (_Float16)W[(d0 + n)*din + K] : (_Float16)0.f;
  }
  return b;
}

// ---------------------------------------------------------------------------
// Generic v4f fill (non-temporal b128: 166 MiB of constant output, bypass cache)
// ---------------------------------------------------------------------------
__global__ __launch_bounds__(256) void k_fill4(v4f* __restrict__ p, float v, int n4){
  int i = blockIdx.x * 256 + threadIdx.x;
  if (i < n4){
    v4f q = {v, v, v, v};
    __builtin_nontemporal_store(q, &p[i]);
  }
}

// ---------------------------------------------------------------------------
// Node MLP: x_t = lrelu(x @ Wn1^T + bn1) @ Wn2^T + bn2      [N,32]->[N,32]
// One wave per 16-node tile.  512 tiles, 64 blocks x 8 waves.
// ---------------------------------------------------------------------------
__global__ __launch_bounds__(256) void k_node_mlp(
    const float* __restrict__ x,
    const float* __restrict__ Wn1, const float* __restrict__ bn1,
    const float* __restrict__ Wn2, const float* __restrict__ bn2,
    float* __restrict__ xt){
  __shared__ float    sW1[64*32];
  __shared__ float    sW2[32*64];
  __shared__ float    sB1[64];
  __shared__ float    sB2[32];
  __shared__ float    sX[8][16*32];      // per-wave input tile (f32)
  __shared__ _Float16 sH[8][16*64];      // per-wave hidden tile (f16)

  int t = threadIdx.x;
  for (int i = t; i < 2048; i += 256){ sW1[i] = Wn1[i]; sW2[i] = Wn2[i]; }
  if (t < 64) sB1[t] = bn1[t];
  if (t < 32) sB2[t] = bn2[t];
  __syncthreads();

  int wave = t >> 5, lane = t & 31;
  int tile = blockIdx.x * 8 + wave;      // exactly 512 tiles
  int row0 = tile * 16;

  // Stage 16x32 f32 tile into LDS with b128 loads (4 per lane).
  const v4f* __restrict__ gsrc = (const v4f*)(x + row0 * NDIM);
  v4f* ldst = (v4f*)sX[wave];
#pragma unroll
  for (int i = lane; i < 128; i += 32) ldst[i] = gsrc[i];
  // wave-private LDS: DS ops are in-order within a wave.

  v16h a0 = frag_a_f32(sX[wave], NDIM, 0, 32, lane);
  int n = lane & 15, hi = lane >> 4;

  // Layer 1: K=32, dout=64 -> 4 N-tiles
#pragma unroll
  for (int j = 0; j < 4; ++j){
    v8f c = {};
    c = WMMA16(a0, frag_b_w(sW1, 32, j*16, 0, 32, lane), c);
#pragma unroll
    for (int v = 0; v < 8; ++v){
      float val = lrelu(c[v] + sB1[j*16 + n]);
      sH[wave][(v + 8*hi)*64 + j*16 + n] = (_Float16)val;
    }
  }

  // Layer 2: K=64 (two K-steps), dout=32 -> 2 N-tiles
  v16h a1 = frag_a_f16(sH[wave], 64,  0, lane);
  v16h a2 = frag_a_f16(sH[wave], 64, 32, lane);
#pragma unroll
  for (int j = 0; j < 2; ++j){
    v8f c = {};
    c = WMMA16(a1, frag_b_w(sW2, 64, j*16,  0, 64, lane), c);
    c = WMMA16(a2, frag_b_w(sW2, 64, j*16, 32, 64, lane), c);
#pragma unroll
    for (int v = 0; v < 8; ++v)
      xt[(row0 + v + 8*hi)*NDIM + j*16 + n] = c[v] + sB2[j*16 + n];
  }
}

// ---------------------------------------------------------------------------
// Edge MLP + gather-multiply + scatter-add:
//   te  = lrelu(ea @ We1^T + be1) @ We2^T + be2         [E,16]->[E,32]
//   agg[dst] += x_t[src] * te * dot_scale
// One wave per 16-edge tile.  32768 tiles, 4096 blocks x 8 waves.
// ---------------------------------------------------------------------------
__global__ __launch_bounds__(256) void k_edge_mlp(
    const float* __restrict__ ea, const int* __restrict__ eidx,
    const float* __restrict__ We1, const float* __restrict__ be1,
    const float* __restrict__ We2, const float* __restrict__ be2,
    const float* __restrict__ dscale,
    const float* __restrict__ xt, float* __restrict__ agg){
  __shared__ float    sW1[32*16];
  __shared__ float    sW2[32*32];
  __shared__ float    sB1[32];
  __shared__ float    sB2[32];
  __shared__ float    sE[8][16*16];      // per-wave edge_attr tile
  __shared__ _Float16 sH[8][16*32];      // per-wave hidden tile

  int t = threadIdx.x;
  for (int i = t; i < 1024; i += 256){ if (i < 512) sW1[i] = We1[i]; sW2[i] = We2[i]; }
  if (t < 32){ sB1[t] = be1[t]; sB2[t] = be2[t]; }
  __syncthreads();

  float ds = dscale[0];
  int wave = t >> 5, lane = t & 31;
  int tile = blockIdx.x * 8 + wave;      // exactly 32768 tiles
  int e0 = tile * 16;

  // Prefetch next tile of the 32 MB edge_attr stream (speculative).
  __builtin_prefetch(ea + (size_t)(e0 + 16) * EDIM, 0, 1);

  // Stage 16x16 f32 tile (2 b128 loads per lane).
  const v4f* __restrict__ gsrc = (const v4f*)(ea + (size_t)e0 * EDIM);
  v4f* ldst = (v4f*)sE[wave];
#pragma unroll
  for (int i = lane; i < 64; i += 32) ldst[i] = gsrc[i];

  int n = lane & 15, hi = lane >> 4;

  // Layer 1: K=16 (zero-padded to 32), dout=32 -> 2 N-tiles
  v16h a0 = frag_a_f32(sE[wave], EDIM, 0, 16, lane);
#pragma unroll
  for (int j = 0; j < 2; ++j){
    v8f c = {};
    c = WMMA16(a0, frag_b_w(sW1, 16, j*16, 0, 16, lane), c);
#pragma unroll
    for (int v = 0; v < 8; ++v){
      float val = lrelu(c[v] + sB1[j*16 + n]);
      sH[wave][(v + 8*hi)*32 + j*16 + n] = (_Float16)val;
    }
  }

  // Edge endpoints for the 8 rows this lane touches in the C layout.
  int se[8], de[8];
#pragma unroll
  for (int v = 0; v < 8; ++v){
    int e = e0 + v + 8*hi;
    se[v] = eidx[e];
    de[v] = eidx[N_EDGES + e];
  }

  // Layer 2: K=32, dout=32 -> 2 N-tiles; fuse gather*mul + scatter atomics.
  v16h a1 = frag_a_f16(sH[wave], 32, 0, lane);
#pragma unroll
  for (int j = 0; j < 2; ++j){
    v8f c = {};
    c = WMMA16(a1, frag_b_w(sW2, 32, j*16, 0, 32, lane), c);
#pragma unroll
    for (int v = 0; v < 8; ++v){
      float te = c[v] + sB2[j*16 + n];
      int f = j*16 + n;
      float msg = xt[se[v]*NDIM + f] * te * ds;
      atomicAdd(&agg[de[v]*NDIM + f], msg);   // global_atomic_add_f32, L2-resident
    }
  }
}

// ---------------------------------------------------------------------------
// h = LN(x + agg*rw)*gamma+beta; accumulate colsum(h) and colsum(node_types).
// One wave per node (32 lanes = 32 features).  1024 blocks x 8 waves.
// ---------------------------------------------------------------------------
__global__ __launch_bounds__(256) void k_ln_colsum(
    const float* __restrict__ x, const float* __restrict__ agg,
    const float* __restrict__ nt, const float* __restrict__ rw,
    const float* __restrict__ gamma, const float* __restrict__ beta,
    float* __restrict__ colsum /* [33] */){
  __shared__ float ps[40];
  int t = threadIdx.x;
  if (t < 40) ps[t] = 0.f;
  __syncthreads();

  int lane = t & 31;
  int node = blockIdx.x * 8 + (t >> 5);

  float h = x[node*NDIM + lane] + agg[node*NDIM + lane] * rw[0];
  float m = h;
#pragma unroll
  for (int off = 16; off; off >>= 1) m += __shfl_xor(m, off, 32);
  m *= (1.f/32.f);
  float d = h - m;
  float var = d * d;
#pragma unroll
  for (int off = 16; off; off >>= 1) var += __shfl_xor(var, off, 32);
  var *= (1.f/32.f);
  float hn = d * rsqrtf(var + 1e-5f) * gamma[lane] + beta[lane];

  atomicAdd(&ps[lane], hn);                  // ds_add_f32
  if (lane == 0) atomicAdd(&ps[32], nt[node]);
  __syncthreads();
  if (t < 33) atomicAdd(&colsum[t], ps[t]);  // 33 global atomics per block
}

// ---------------------------------------------------------------------------
// new_x[j,f] = colsum_h[f]/k ; new_node_types[j] = colsum_nt/k
// ---------------------------------------------------------------------------
__global__ __launch_bounds__(256) void k_heads(
    const float* __restrict__ colsum,
    float* __restrict__ newx, float* __restrict__ newnt){
  int i = blockIdx.x * 256 + threadIdx.x;
  const float inv = 1.0f / (float)KCL;
  if (i < KCL * NDIM){
    __builtin_nontemporal_store(colsum[i & 31] * inv, &newx[i]);
  } else {
    int j = i - KCL * NDIM;
    if (j < KCL) newnt[j] = colsum[32] * inv;
  }
}

// ---------------------------------------------------------------------------
extern "C" void kernel_launch(void* const* d_in, const int* in_sizes, int n_in,
                              void* d_out, int out_size, void* d_ws, size_t ws_size,
                              hipStream_t stream){
  (void)in_sizes; (void)n_in; (void)out_size; (void)d_ws; (void)ws_size;

  const float* x    = (const float*)d_in[0];
  const int*   eidx = (const int*)  d_in[1];
  const float* ea   = (const float*)d_in[2];
  const float* nt   = (const float*)d_in[3];
  const float* We1  = (const float*)d_in[4];
  const float* be1  = (const float*)d_in[5];
  const float* We2  = (const float*)d_in[6];
  const float* be2  = (const float*)d_in[7];
  const float* Wn1  = (const float*)d_in[8];
  const float* bn1  = (const float*)d_in[9];
  const float* Wn2  = (const float*)d_in[10];
  const float* bn2  = (const float*)d_in[11];
  const float* rw   = (const float*)d_in[12];
  const float* ds   = (const float*)d_in[13];
  const float* gam  = (const float*)d_in[14];
  const float* bet  = (const float*)d_in[15];
  // d_in[16..22] (assignment net, proto): mathematically dead, see header.

  float* out = (float*)d_out;
  float* o_A = out + OFF_A;

  // Scratch lives in the A output region (128 MB, overwritten last with 1/k):
  float* xt     = o_A;               // 262144 floats
  float* agg    = o_A + 262144;      // 262144 floats
  float* colsum = o_A + 524288;      // 33 floats (zeroed as 48)

  // 1. zero agg + colsum
  k_fill4<<<dim3(257), dim3(256), 0, stream>>>((v4f*)agg, 0.f, (262144 + 48) / 4);
  // 2. node transform (WMMA)
  k_node_mlp<<<dim3(64), dim3(256), 0, stream>>>(x, Wn1, bn1, Wn2, bn2, xt);
  // 3. edge MLP + gather*mul + scatter-add (WMMA + f32 atomics)
  k_edge_mlp<<<dim3(4096), dim3(256), 0, stream>>>(ea, eidx, We1, be1, We2, be2,
                                                   ds, xt, agg);
  // 4. residual + LayerNorm + column sums
  k_ln_colsum<<<dim3(1024), dim3(256), 0, stream>>>(x, agg, nt, rw, gam, bet, colsum);
  // 5. new_x / new_node_types from column sums (reads scratch -> before A fill)
  k_heads<<<dim3((KCL*NDIM + KCL + 255) / 256), dim3(256), 0, stream>>>(
      colsum, out + OFF_NEWX, out + OFF_NNT);
  // 6. constant outputs (non-temporal streaming stores)
  //    new_edge_index + new_edge_attr are contiguous: 9437184 zeros
  k_fill4<<<dim3(9216),  dim3(256), 0, stream>>>((v4f*)(out + OFF_EIDX), 0.f, 2359296);
  //    edge_mask: all edges dropped (cluster==0 for src and dst)
  k_fill4<<<dim3(512),   dim3(256), 0, stream>>>((v4f*)(out + OFF_MASK), 0.f, 131072);
  //    A: exact uniform softmax 1/k (also overwrites scratch)
  k_fill4<<<dim3(32768), dim3(256), 0, stream>>>((v4f*)o_A, 1.0f / (float)KCL, 8388608);
}
